// SoftHistogram_47399259079035
// MI455X (gfx1250) — matrix operand order repaired and checked by reading, not compiled, not run on hardware
//
#include <hip/hip_runtime.h>
#include <hip/hip_bf16.h>

typedef __attribute__((ext_vector_type(16))) _Float16 v16h;
typedef __attribute__((ext_vector_type(8)))  float    v8f;

#define NPIX   147456          // 384*384
#define NTOT   1179648         // 8*384*384
#define NCH    36              // histogram chunks per batch
#define CHSZ   4096            // elements per chunk (36*4096 == 147456)
#define NTILE  576             // 24*24 16x16 tiles per image
#define NBLK_SSIM (8*NTILE)

// 11-tap Gaussian, sigma=1.5, normalized (matches reference _gauss_window)
__constant__ float c_G[11] = {
    0.00102838f, 0.00759877f, 0.03600079f, 0.10936045f, 0.21300539f,
    0.26601182f,
    0.21300539f, 0.10936045f, 0.03600079f, 0.00759877f, 0.00102838f};

__device__ __forceinline__ float sigmoidf_fast(float z) {
    // sigmoid(z) = 1 / (1 + 2^(-z*log2(e)))   -> v_exp_f32 + v_rcp_f32
    float u = __builtin_amdgcn_exp2f(z * -1.44269504088896340736f);
    return __builtin_amdgcn_rcpf(1.0f + u);
}

// op: 0=add, 1=min, 2=max ; blockDim.x must be 256
__device__ float bred_op(float v, float* s, int op) {
    int tid = threadIdx.x;
    s[tid] = v;
    __syncthreads();
    for (int off = 128; off > 0; off >>= 1) {
        if (tid < off) {
            float a = s[tid], b = s[tid + off];
            s[tid] = (op == 0) ? (a + b) : (op == 1) ? fminf(a, b) : fmaxf(a, b);
        }
        __syncthreads();
    }
    float r = s[0];
    __syncthreads();
    return r;
}

// ---------------------------------------------------------------------------
// K1: per-block partial min/max of x,target,depth + sums (x, d, x*d)
// ---------------------------------------------------------------------------
__global__ void __launch_bounds__(256)
k_reduce1(const float* __restrict__ x, const float* __restrict__ t,
          const float* __restrict__ d, int n, float* __restrict__ part) {
    float xmn = 3.0e38f, xmx = -3.0e38f;
    float tmn = 3.0e38f, tmx = -3.0e38f;
    float dmn = 3.0e38f, dmx = -3.0e38f;
    float sx = 0.f, sd = 0.f, sxd = 0.f;
    for (int i = blockIdx.x * blockDim.x + threadIdx.x; i < n;
         i += gridDim.x * blockDim.x) {
        float xv = x[i], tv = t[i], dv = d[i];
        xmn = fminf(xmn, xv); xmx = fmaxf(xmx, xv);
        tmn = fminf(tmn, tv); tmx = fmaxf(tmx, tv);
        dmn = fminf(dmn, dv); dmx = fmaxf(dmx, dv);
        sx += xv; sd += dv; sxd += xv * dv;
    }
    __shared__ float s[256];
    float r[9];
    r[0] = bred_op(xmn, s, 1); r[1] = bred_op(xmx, s, 2);
    r[2] = bred_op(tmn, s, 1); r[3] = bred_op(tmx, s, 2);
    r[4] = bred_op(dmn, s, 1); r[5] = bred_op(dmx, s, 2);
    r[6] = bred_op(sx, s, 0);  r[7] = bred_op(sd, s, 0);
    r[8] = bred_op(sxd, s, 0);
    if (threadIdx.x == 0)
        for (int q = 0; q < 9; ++q) part[blockIdx.x * 9 + q] = r[q];
}

// ---------------------------------------------------------------------------
// K2: fold the 256 partials into 9 scalars: min / 1/range per tensor + sums
// ---------------------------------------------------------------------------
__global__ void __launch_bounds__(256)
k_scalars(const float* __restrict__ part, float* __restrict__ scal) {
    __shared__ float s[256];
    int tid = threadIdx.x;
    float p[9];
    for (int q = 0; q < 9; ++q) p[q] = part[tid * 9 + q];
    float xmn = bred_op(p[0], s, 1), xmx = bred_op(p[1], s, 2);
    float tmn = bred_op(p[2], s, 1), tmx = bred_op(p[3], s, 2);
    float dmn = bred_op(p[4], s, 1), dmx = bred_op(p[5], s, 2);
    float sx = bred_op(p[6], s, 0), sd = bred_op(p[7], s, 0);
    float sxd = bred_op(p[8], s, 0);
    if (tid == 0) {
        scal[0] = xmn; scal[1] = 1.0f / (xmx - xmn);
        scal[2] = tmn; scal[3] = 1.0f / (tmx - tmn);
        scal[4] = dmn; scal[5] = 1.0f / (dmx - dmn);
        scal[6] = sx;  scal[7] = sd;  scal[8] = sxd;
    }
}

// ---------------------------------------------------------------------------
// K3: soft histogram. One thread per bin; sigmoid shared between adjacent
// bins via wave32 shfl_down (lane 31 recomputes the wave-boundary edge).
// ---------------------------------------------------------------------------
__global__ void __launch_bounds__(256)
k_hist(const float* __restrict__ x, const float* __restrict__ scal,
       float* __restrict__ histp) {
    const int b = blockIdx.x / NCH, ch = blockIdx.x % NCH;
    const int tid = threadIdx.x;
    const float xm = scal[0], xi = scal[1];
    const float SIG = 3.0f, DELTA = 1.0f / 256.0f;
    const float edge0 = tid * DELTA;
    const float edge1 = (tid + 1) * DELTA;
    const bool last = ((tid & 31) == 31);
    const float* px = x + (size_t)b * NPIX + (size_t)ch * CHSZ;

    __shared__ float xe[256];
    float h = 0.f;
    for (int base = 0; base < CHSZ; base += 256) {
        xe[tid] = (px[base + tid] - xm) * xi;
        __syncthreads();
        for (int j = 0; j < 256; ++j) {
            float e  = xe[j];
            float s0 = sigmoidf_fast(SIG * (e - edge0));
            float s1 = __shfl_down(s0, 1, 32);
            if (last) s1 = sigmoidf_fast(SIG * (e - edge1));
            h += s0 - s1;
        }
        __syncthreads();
    }
    histp[blockIdx.x * 256 + tid] = h;
}

// ---------------------------------------------------------------------------
// K4: SSIM via WMMA. One wave per 16x16 output tile. Separable 11-tap
// Gaussian conv expressed as banded-matrix matmuls:
//   horizontal: D(16x16) = A(img rows 16x32) x Wh(32x16), Wh[k][n] = g[k-n]
//   vertical:   D(16x16) = Wv(16x32) x B(Ht 32x16),       Wv[m][k] = g[k-m]
// f16 operands / f32 accumulate using v_wmma_f32_16x16x32_f16.
// ---------------------------------------------------------------------------
__global__ void __launch_bounds__(32)
k_ssim(const float* __restrict__ x, const float* __restrict__ tg,
       const float* __restrict__ scal, float* __restrict__ ssimp) {
    __shared__ float tA[32 * 32];   // normalized x tile (26x26 live, zero pad)
    __shared__ float tB[32 * 32];   // normalized target tile
    __shared__ float Ht[32 * 16];   // horizontal-pass result (rows 0..25 live)

    const int lane = threadIdx.x;
    const int bid = blockIdx.x;
    const int b = bid / NTILE, tl = bid % NTILE;
    const int ty = tl / 24, tx = tl % 24;
    const int row0 = ty * 16 - 5, col0 = tx * 16 - 5;
    const float xm = scal[0], xi = scal[1], tm = scal[2], ti = scal[3];
    const float* px = x + (size_t)b * NPIX;
    const float* pt = tg + (size_t)b * NPIX;

    for (int i = lane; i < 1024; i += 32) { tA[i] = 0.f; tB[i] = 0.f; }
    __syncthreads();
    for (int i = lane; i < 26 * 26; i += 32) {
        int r = i / 26, c = i % 26;
        int gr = row0 + r, gc = col0 + c;
        if (gr >= 0 && gr < 384 && gc >= 0 && gc < 384) {
            tA[r * 32 + c] = (px[gr * 384 + gc] - xm) * xi;
            tB[r * 32 + c] = (pt[gr * 384 + gc] - tm) * ti;
        }
    }
    __syncthreads();

    const bool hi = lane >= 16;
    const int lm = lane & 15;

    // Banded Gaussian weight matrices, built once in this file's contiguous k order (k = 16*half + elem).
    v16h wB;   // B-layout: lanes<16 K=2v+e, lanes>=16 K=16+2v+e; N=lm
    v16h wA;   // A-layout: lanes<16 K=2v+e, lanes>=16 K=16+2v+e; M=lm
#pragma unroll
    for (int v = 0; v < 8; ++v)
#pragma unroll
        for (int e = 0; e < 2; ++e) {
            int kB = (hi ? 16 : 0) + 2 * v + e;
            int dB = kB - lm;
            wB[2 * v + e] = (_Float16)((dB >= 0 && dB <= 10) ? c_G[dB] : 0.f);
            int kA = (hi ? 16 : 0) + 2 * v + e;
            int dA = kA - lm;
            wA[2 * v + e] = (_Float16)((dA >= 0 && dA <= 10) ? c_G[dA] : 0.f);
        }

    v8f res[5];
#pragma unroll
    for (int fi = 0; fi < 5; ++fi) {
        // --- horizontal pass: 2 row groups of 16 image rows ---
#pragma unroll
        for (int g = 0; g < 2; ++g) {
            v16h a;
            int m = g * 16 + lm;
#pragma unroll
            for (int v = 0; v < 8; ++v)
#pragma unroll
                for (int e = 0; e < 2; ++e) {
                    int k = (hi ? 16 : 0) + 2 * v + e;
                    float x1 = tA[m * 32 + k], x2 = tB[m * 32 + k];
                    float val = (fi == 0) ? x1 : (fi == 1) ? x2
                              : (fi == 2) ? x1 * x1
                              : (fi == 3) ? x2 * x2 : x1 * x2;
                    a[2 * v + e] = (_Float16)val;
                }
            v8f c = {};
            c = __builtin_amdgcn_wmma_f32_16x16x32_f16(
                    false, a, false, wB, (short)0, c, false, false);
            int mr = hi ? 8 : 0;
#pragma unroll
            for (int r = 0; r < 8; ++r)
                Ht[(g * 16 + r + mr) * 16 + lm] = c[r];
        }
        __syncthreads();
        // --- vertical pass ---
        v16h bv;
#pragma unroll
        for (int v = 0; v < 8; ++v)
#pragma unroll
            for (int e = 0; e < 2; ++e) {
                int k = (hi ? 16 : 0) + 2 * v + e;
                bv[2 * v + e] = (_Float16)Ht[k * 16 + lm];
            }
        v8f c = {};
        c = __builtin_amdgcn_wmma_f32_16x16x32_f16(
                false, wA, false, bv, (short)0, c, false, false);
        res[fi] = c;
        __syncthreads();
    }

    const float C1 = 1e-4f, C2 = 9e-4f;
    float acc = 0.f;
#pragma unroll
    for (int r = 0; r < 8; ++r) {
        float mu1 = res[0][r], mu2 = res[1][r];
        float exx = res[2][r], eyy = res[3][r], exy = res[4][r];
        float m11 = mu1 * mu1, m22 = mu2 * mu2, m12 = mu1 * mu2;
        float s1 = exx - m11, s2 = eyy - m22, s12 = exy - m12;
        float num = (2.f * m12 + C1) * (2.f * s12 + C2);
        float den = (m11 + m22 + C1) * (s1 + s2 + C2);
        acc += num / den;
    }
#pragma unroll
    for (int off = 16; off > 0; off >>= 1) acc += __shfl_down(acc, off, 32);
    if (lane == 0) ssimp[bid] = acc;
}

// ---------------------------------------------------------------------------
// K5: combine everything into the scalar loss
// ---------------------------------------------------------------------------
__global__ void __launch_bounds__(256)
k_final(const float* __restrict__ histp, const float* __restrict__ ssimp,
        const float* __restrict__ scal, float* __restrict__ out) {
    __shared__ float s[256];
    const int tid = threadIdx.x;
    // uniform loss: thread tid owns bin tid across all batches
    float acc = 0.f;
    for (int b = 0; b < 8; ++b) {
        float h = 0.f;
        for (int ch = 0; ch < NCH; ++ch)
            h += histp[(b * NCH + ch) * 256 + tid];
        // natural log via v_log_f32 (base-2) * ln(2)
        if (h > 0.f)
            acc += h * (__builtin_amdgcn_logf(h) * 0.69314718055994530942f);
    }
    float uni = bred_op(acc, s, 0);
    // ssim partial tile sums
    float sa = 0.f;
    for (int i = tid; i < NBLK_SSIM; i += 256) sa += ssimp[i];
    float ssum = bred_op(sa, s, 0);
    if (tid == 0) {
        float uniform_loss = uni / 8.f;
        float ssim_loss = 1.f - ssum / (float)NTOT;
        float xm = scal[0], xi = scal[1], dm = scal[4], di = scal[5];
        float sx = scal[6], sd = scal[7], sxd = scal[8];
        float N = (float)NTOT;
        float dsum = (sxd - dm * sx - xm * sd + N * xm * dm) * xi * di;
        float depth_loss = dsum / 8.f;
        out[0] = 1e-6f * uniform_loss + 1e-5f * depth_loss + ssim_loss;
    }
}

extern "C" void kernel_launch(void* const* d_in, const int* in_sizes, int n_in,
                              void* d_out, int out_size, void* d_ws,
                              size_t ws_size, hipStream_t stream) {
    (void)in_sizes; (void)n_in; (void)out_size; (void)ws_size;
    const float* x  = (const float*)d_in[0];   // [8,1,384,384]
    const float* tg = (const float*)d_in[1];   // [8,384,384]
    const float* dp = (const float*)d_in[2];   // [8,384,384]
    float* ws = (float*)d_ws;
    float* scal  = ws;                 // 16 floats
    float* part  = ws + 16;            // 256*9
    float* histp = ws + 4096;          // 8*36*256 = 73728
    float* ssimp = ws + 4096 + 73728;  // 4608

    k_reduce1<<<256, 256, 0, stream>>>(x, tg, dp, NTOT, part);
    k_scalars<<<1, 256, 0, stream>>>(part, scal);
    k_hist<<<8 * NCH, 256, 0, stream>>>(x, scal, histp);
    k_ssim<<<NBLK_SSIM, 32, 0, stream>>>(x, tg, scal, ssimp);
    k_final<<<1, 256, 0, stream>>>(histp, ssimp, scal, (float*)d_out);
}
